// SAModule_34892314313494
// MI455X (gfx1250) — compile-verified, matrix operand order checked by reading, not run on hardware
//
#include <hip/hip_runtime.h>

typedef __attribute__((ext_vector_type(16))) _Float16 v16h;
typedef __attribute__((ext_vector_type(8)))  _Float16 v8h;
typedef __attribute__((ext_vector_type(4)))  _Float16 v4h;
typedef __attribute__((ext_vector_type(8)))  float    v8f;

#define Bn    16
#define Nn    4096
#define Fn    64
#define Mn    1024
#define Kn    64
#define CINP  96          // 67 padded to 3 k-steps of 32
#define H1n   64
#define H2n   64
#define OUTn  128
#define RAD2  0.04f       // R^2
#define NEGI  -3.0e38f
#define CANDMAX 2048

// ws layout: [0, 36864) bytes = swizzled f16 weights; nbr ints after.
#define W1_OFF 0
#define W2_OFF 6144       // f16 elements
#define W3_OFF 10240
#define W_TOTAL 18432     // f16 elements (36864 bytes)

// ---------------------------------------------------------------------------
// WMMA helpers (CDNA5 v_wmma_f32_16x16x32_f16)
// ---------------------------------------------------------------------------
__device__ inline v8f wmma16(v16h a, v16h b, v8f c) {
  return __builtin_amdgcn_wmma_f32_16x16x32_f16(false, a, false, b,
                                                (short)0, c, false, false);
}

// A fragment 16x32 (MxK) from row-major LDS [16][ld]:
// lane's 16 values are two contiguous 8-f16 runs -> two ds_load_b128.
__device__ inline v16h load_a(const _Float16* src, int ld, int lane) {
  const int row = lane & 15;
  const int kh  = (lane >> 4) & 1;
  const _Float16* rp = src + row * ld + kh * 8;
  v8h lo = *(const v8h*)(rp);
  v8h hi = *(const v8h*)(rp + 16);
  return __builtin_shufflevector(lo, hi, 0, 1, 2, 3, 4, 5, 6, 7,
                                 8, 9, 10, 11, 12, 13, 14, 15);
}

// B fragment from fragment-major (pre-swizzled) storage: lane's 16 f16 are
// fully contiguous -> two b128 loads (works from global; stays hot in WGP$).
__device__ inline v16h load_b_packed(const _Float16* fragbase, int lane) {
  const _Float16* p = fragbase + lane * 16;
  v8h lo = *(const v8h*)(p);
  v8h hi = *(const v8h*)(p + 8);
  return __builtin_shufflevector(lo, hi, 0, 1, 2, 3, 4, 5, 6, 7,
                                 8, 9, 10, 11, 12, 13, 14, 15);
}

// (k,n) of a KxN weight -> element index in fragment-major swizzled layout
__device__ inline int swz_idx(int k, int n, int Ndim) {
  const int ks = k >> 5, kl = k & 31;
  int kh, j;
  if (kl < 16) { kh = kl >> 3; j = kl & 7; }
  else         { kh = (kl - 16) >> 3; j = 8 + ((kl - 16) & 7); }
  const int lane = (n & 15) + (kh << 4);
  const int frag = ks * (Ndim >> 4) + (n >> 4);
  return frag * 512 + lane * 16 + j;
}

// ---------------------------------------------------------------------------
// Kernel 0: one-shot weight prep (f32 -> f16, fragment-major swizzle)
// ---------------------------------------------------------------------------
__global__ void prep_kernel(const float* __restrict__ W1,
                            const float* __restrict__ W2,
                            const float* __restrict__ W3,
                            _Float16* __restrict__ ws) {
  const int i = blockIdx.x * 256 + threadIdx.x;
  if (i < 6144) {                              // W1: 96(pad from 67) x 64
    const int k = i >> 6, n = i & 63;
    const float v = (k < Fn + 3) ? W1[k * 64 + n] : 0.0f;
    ws[W1_OFF + swz_idx(k, n, 64)] = (_Float16)v;
  } else if (i < 10240) {                      // W2: 64 x 64
    const int t = i - 6144;
    ws[W2_OFF + swz_idx(t >> 6, t & 63, 64)] = (_Float16)W2[t];
  } else if (i < W_TOTAL) {                    // W3: 64 x 128
    const int t = i - 10240;
    ws[W3_OFF + swz_idx(t >> 7, t & 127, 128)] = (_Float16)W3[t];
  }
}

// ---------------------------------------------------------------------------
// Kernel 1: farthest point sampling.  1 block (8 waves) per cloud.
// Shfl-based argmax: 2 barriers per sequential step.
// ---------------------------------------------------------------------------
__global__ void fps_kernel(const float* __restrict__ pos,
                           float* __restrict__ pos_s) {
  const int b    = blockIdx.x;
  const int tid  = threadIdx.x;
  const int lane = tid & 31;
  const int wid  = tid >> 5;
  const float* P = pos + (size_t)b * Nn * 3;

  __shared__ float sWv[8];
  __shared__ int   sWi[8];
  __shared__ int   sLast;

  float dmin[16];
#pragma unroll
  for (int t = 0; t < 16; ++t) dmin[t] = 1e10f;

  int last = 0;
  for (int m = 0; m < Mn; ++m) {
    if (tid == 0) {
      float* o = pos_s + (size_t)(b * Mn + m) * 3;
      o[0] = P[last * 3 + 0];
      o[1] = P[last * 3 + 1];
      o[2] = P[last * 3 + 2];
    }
    const float lx = P[last * 3 + 0];
    const float ly = P[last * 3 + 1];
    const float lz = P[last * 3 + 2];

    float bv = -1.0f; int bi = 0;
#pragma unroll
    for (int t = 0; t < 16; ++t) {
      const int p = tid + t * 256;
      const float dx = P[p * 3 + 0] - lx;
      const float dy = P[p * 3 + 1] - ly;
      const float dz = P[p * 3 + 2] - lz;
      const float d  = dx * dx + dy * dy + dz * dz;
      dmin[t] = fminf(dmin[t], d);
      if (dmin[t] > bv) { bv = dmin[t]; bi = p; }
    }
#pragma unroll
    for (int off = 16; off > 0; off >>= 1) {
      const float ov = __shfl_xor(bv, off, 32);
      const int   oi = __shfl_xor(bi, off, 32);
      if (ov > bv) { bv = ov; bi = oi; }
    }
    if (lane == 0) { sWv[wid] = bv; sWi[wid] = bi; }
    __syncthreads();
    if (wid == 0) {
      float v  = (lane < 8) ? sWv[lane] : -1.0f;
      int   ii = (lane < 8) ? sWi[lane] : 0;
#pragma unroll
      for (int off = 4; off > 0; off >>= 1) {
        const float ov = __shfl_xor(v, off, 32);
        const int   oi = __shfl_xor(ii, off, 32);
        if (ov > v) { v = ov; ii = oi; }
      }
      if (lane == 0) sLast = ii;
    }
    __syncthreads();
    last = sLast;
  }
}

// ---------------------------------------------------------------------------
// Kernel 2: ball query.  Compact in-radius candidates once, then K argmin
// rounds over the compact list (shfl reductions, 2 barriers per round).
// ---------------------------------------------------------------------------
__global__ void ball_kernel(const float* __restrict__ pos,
                            const float* __restrict__ pos_s,
                            int* __restrict__ nbr) {
  const int cid  = blockIdx.x;
  const int b    = cid >> 10;
  const int tid  = threadIdx.x;
  const int lane = tid & 31;
  const int wid  = tid >> 5;
  const float* P = pos + (size_t)b * Nn * 3;

  __shared__ int   sCand[CANDMAX];
  __shared__ float sCd[CANDMAX];
  __shared__ int   sCnt;
  __shared__ float sWv[8];
  __shared__ int   sWi[8];
  __shared__ float sBv;
  __shared__ int   sBi;

  if (tid == 0) sCnt = 0;
  __syncthreads();

  const float cx = pos_s[(size_t)cid * 3 + 0];
  const float cy = pos_s[(size_t)cid * 3 + 1];
  const float cz = pos_s[(size_t)cid * 3 + 2];

  for (int p = tid; p < Nn; p += 256) {
    const float dx = P[p * 3 + 0] - cx;
    const float dy = P[p * 3 + 1] - cy;
    const float dz = P[p * 3 + 2] - cz;
    const float d2 = dx * dx + dy * dy + dz * dz;
    if (d2 <= RAD2) {
      const int slot = atomicAdd(&sCnt, 1);
      if (slot < CANDMAX) { sCand[slot] = p; sCd[slot] = d2; }
    }
  }
  __syncthreads();
  const int ncand = (sCnt < CANDMAX) ? sCnt : CANDMAX;

  for (int k = 0; k < Kn; ++k) {
    float bv = 3.0e38f; int bi = -1;
    for (int i = tid; i < ncand; i += 256) {
      const float d = sCd[i];
      if (d < bv) { bv = d; bi = i; }
    }
#pragma unroll
    for (int off = 16; off > 0; off >>= 1) {
      const float ov = __shfl_xor(bv, off, 32);
      const int   oi = __shfl_xor(bi, off, 32);
      if (ov < bv) { bv = ov; bi = oi; }
    }
    if (lane == 0) { sWv[wid] = bv; sWi[wid] = bi; }
    __syncthreads();
    if (wid == 0) {
      float v  = (lane < 8) ? sWv[lane] : 3.0e38f;
      int   ii = (lane < 8) ? sWi[lane] : -1;
#pragma unroll
      for (int off = 4; off > 0; off >>= 1) {
        const float ov = __shfl_xor(v, off, 32);
        const int   oi = __shfl_xor(ii, off, 32);
        if (ov < v) { v = ov; ii = oi; }
      }
      if (lane == 0) {
        const int res = (ii >= 0 && v <= RAD2) ? sCand[ii] : -1;
        nbr[(size_t)cid * Kn + k] = res;
        if (ii >= 0) sCd[ii] = 3.0e38f;     // remove from future rounds
        sBv = v; sBi = ii;
      }
    }
    __syncthreads();
  }
}

// ---------------------------------------------------------------------------
// Kernel 3: gather + 3-layer MLP (WMMA) + masked max.
// 1 block per center, 4 waves; wave w owns neighbor rows [16w, 16w+16).
// B operands come pre-swizzled from global (hot in WGP$/L2).
// ---------------------------------------------------------------------------
__global__ void __launch_bounds__(128)
conv_kernel(const float* __restrict__ x,   const float* __restrict__ pos,
            const float* __restrict__ b1,  const float* __restrict__ b2,
            const float* __restrict__ b3,
            const _Float16* __restrict__ wsW,
            const float* __restrict__ pos_s, const int* __restrict__ nbr,
            float* __restrict__ out) {
  const int cid  = blockIdx.x;
  const int b    = cid >> 10;
  const int tid  = threadIdx.x;
  const int lane = tid & 31;
  const int w    = tid >> 5;             // wave id = M-tile id

  __shared__ _Float16 sFeat[Kn * CINP];  // 64 x 96
  __shared__ _Float16 sH[Kn * H1n];      // 64 x 64 (reused between layers)
  __shared__ float    sValid[Kn];
  __shared__ float    sPart[4 * OUTn];

  const _Float16* wB1 = wsW + W1_OFF;
  const _Float16* wB2 = wsW + W2_OFF;
  const _Float16* wB3 = wsW + W3_OFF;

  if (tid < Kn) sValid[tid] = (nbr[(size_t)cid * Kn + tid] >= 0) ? 1.0f : 0.0f;

  // --- gather: concat(x_j, pos_j - pos_i), zero-padded; float4 -> v4h ---
  const float cx = pos_s[(size_t)cid * 3 + 0];
  const float cy = pos_s[(size_t)cid * 3 + 1];
  const float cz = pos_s[(size_t)cid * 3 + 2];
  for (int i = tid; i < Kn * (CINP / 4); i += 128) {
    const int r  = i / (CINP / 4);
    const int c0 = (i % (CINP / 4)) * 4;
    int j = nbr[(size_t)cid * Kn + r];
    if (j < 0) j = 0;                    // garbage ok: row masked at max
    v4h h;
    if (c0 < Fn) {
      const float4 v = *(const float4*)&x[((size_t)b * Nn + j) * Fn + c0];
      h[0] = (_Float16)v.x; h[1] = (_Float16)v.y;
      h[2] = (_Float16)v.z; h[3] = (_Float16)v.w;
    } else if (c0 == Fn) {
      const float* pj = &pos[((size_t)b * Nn + j) * 3];
      h[0] = (_Float16)(pj[0] - cx);
      h[1] = (_Float16)(pj[1] - cy);
      h[2] = (_Float16)(pj[2] - cz);
      h[3] = (_Float16)0.0f;
    } else {
      h[0] = h[1] = h[2] = h[3] = (_Float16)0.0f;
    }
    *(v4h*)&sFeat[r * CINP + c0] = h;
  }
  __syncthreads();

  const int half = lane >> 4;
  const int nidx = lane & 15;

  // ---------------- layer 1: [16x96] @ [96x64] + b1, relu ----------------
  {
    v16h a0 = load_a(&sFeat[(w * 16) * CINP +  0], CINP, lane);
    v16h a1 = load_a(&sFeat[(w * 16) * CINP + 32], CINP, lane);
    v16h a2 = load_a(&sFeat[(w * 16) * CINP + 64], CINP, lane);
#pragma unroll
    for (int nt = 0; nt < 4; ++nt) {
      v8f c = {};
      c = wmma16(a0, load_b_packed(wB1 + (0 * 4 + nt) * 512, lane), c);
      c = wmma16(a1, load_b_packed(wB1 + (1 * 4 + nt) * 512, lane), c);
      c = wmma16(a2, load_b_packed(wB1 + (2 * 4 + nt) * 512, lane), c);
      const int col = nt * 16 + nidx;
      const float bias = b1[col];
#pragma unroll
      for (int g = 0; g < 8; ++g) {
        const int r = w * 16 + half * 8 + g;
        const float v = c[g] + bias;
        sH[r * H1n + col] = (_Float16)(v > 0.0f ? v : 0.0f);
      }
    }
  }

  // ---------------- layer 2: [16x64] @ [64x64] + b2, relu ----------------
  {
    v16h a0 = load_a(&sH[(w * 16) * H1n +  0], H1n, lane);  // hoisted before
    v16h a1 = load_a(&sH[(w * 16) * H1n + 32], H1n, lane);  // overwriting sH
#pragma unroll
    for (int nt = 0; nt < 4; ++nt) {
      v8f c = {};
      c = wmma16(a0, load_b_packed(wB2 + (0 * 4 + nt) * 512, lane), c);
      c = wmma16(a1, load_b_packed(wB2 + (1 * 4 + nt) * 512, lane), c);
      const int col = nt * 16 + nidx;
      const float bias = b2[col];
#pragma unroll
      for (int g = 0; g < 8; ++g) {
        const int r = w * 16 + half * 8 + g;
        const float v = c[g] + bias;
        sH[r * H2n + col] = (_Float16)(v > 0.0f ? v : 0.0f);
      }
    }
  }

  // -------- layer 3: [16x64] @ [64x128] + b3, masked max over rows --------
  {
    v16h a0 = load_a(&sH[(w * 16) * H2n +  0], H2n, lane);
    v16h a1 = load_a(&sH[(w * 16) * H2n + 32], H2n, lane);
#pragma unroll
    for (int nt = 0; nt < 8; ++nt) {
      v8f c = {};
      c = wmma16(a0, load_b_packed(wB3 + (0 * 8 + nt) * 512, lane), c);
      c = wmma16(a1, load_b_packed(wB3 + (1 * 8 + nt) * 512, lane), c);
      const int col = nt * 16 + nidx;
      const float bias = b3[col];
      float mx = NEGI;
#pragma unroll
      for (int g = 0; g < 8; ++g) {
        const int r = w * 16 + half * 8 + g;
        const float v = c[g] + bias;
        mx = fmaxf(mx, (sValid[r] != 0.0f) ? v : NEGI);
      }
      mx = fmaxf(mx, __shfl_xor(mx, 16, 32));   // merge the two row-halves
      if (half == 0) sPart[w * OUTn + col] = mx;
    }
  }
  __syncthreads();

  // cross-wave max, write output
  if (tid < OUTn) {
    float m = sPart[0 * OUTn + tid];
    m = fmaxf(m, sPart[1 * OUTn + tid]);
    m = fmaxf(m, sPart[2 * OUTn + tid]);
    m = fmaxf(m, sPart[3 * OUTn + tid]);
    out[(size_t)cid * OUTn + tid] = m;
  }
}

// ---------------------------------------------------------------------------
extern "C" void kernel_launch(void* const* d_in, const int* in_sizes, int n_in,
                              void* d_out, int out_size, void* d_ws, size_t ws_size,
                              hipStream_t stream) {
  const float* x   = (const float*)d_in[0];
  const float* pos = (const float*)d_in[1];
  const float* W1  = (const float*)d_in[2];
  const float* b1  = (const float*)d_in[3];
  const float* W2  = (const float*)d_in[4];
  const float* b2  = (const float*)d_in[5];
  const float* W3  = (const float*)d_in[6];
  const float* b3  = (const float*)d_in[7];

  float* out   = (float*)d_out;                       // [B*M*OUT]
  float* pos_s = out + (size_t)Bn * Mn * OUTn;        // [B*M*3] (2nd output)

  _Float16* wsW = (_Float16*)d_ws;                    // swizzled f16 weights
  int*      nbr = (int*)((char*)d_ws + W_TOTAL * 2);  // [B*M*K]

  prep_kernel<<<W_TOTAL / 256, 256, 0, stream>>>(W1, W2, W3, wsW);
  fps_kernel <<<Bn,            256, 0, stream>>>(pos, pos_s);
  ball_kernel<<<Bn * Mn,       256, 0, stream>>>(pos, pos_s, nbr);
  conv_kernel<<<Bn * Mn,       128, 0, stream>>>(x, pos, b1, b2, b3,
                                                 wsW, pos_s, nbr, out);
}
